// MultiHeadAttention_10187662426636
// MI455X (gfx1250) — matrix operand order repaired and checked by reading, not compiled, not run on hardware
//
#include <hip/hip_runtime.h>

typedef __attribute__((ext_vector_type(16))) __bf16 v16bf;
typedef __attribute__((ext_vector_type(8)))  __bf16 v8bf;
typedef __attribute__((ext_vector_type(8)))  float  v8f;

#define DM   1024
#define NH   16
#define HD   64
#define TS   2048
#define BB   4
#define MR   (BB*TS)   // 8192 rows

__device__ __forceinline__ v16bf cat8(v8bf a, v8bf b) {
  return __builtin_shufflevector(a, b, 0,1,2,3,4,5,6,7,8,9,10,11,12,13,14,15);
}
__device__ __forceinline__ v8f wmma_bf16(v16bf a, v16bf b, v8f c) {
  return __builtin_amdgcn_wmma_f32_16x16x32_bf16(false, a, false, b, (short)0, c, false, false);
}

// ---------------- fp32 -> bf16 conversion ----------------
__global__ void cvt_kernel(const float* __restrict__ in, __bf16* __restrict__ out, int n) {
  int i = blockIdx.x * blockDim.x + threadIdx.x;
  int st = gridDim.x * blockDim.x;
  for (; i < n; i += st) out[i] = (__bf16)in[i];
}

// W[k][n] fp32 -> Wt[n][k] bf16 (so WMMA B-fragments are contiguous row loads)
__global__ void transw_kernel(const float* __restrict__ W, __bf16* __restrict__ Wt) {
  int i = blockIdx.x * blockDim.x + threadIdx.x;
  if (i >= DM * DM) return;
  int nn = i >> 10, kk = i & (DM - 1);
  Wt[(size_t)nn * DM + kk] = (__bf16)W[(size_t)kk * DM + nn];
}

// ---------------- WMMA GEMM: out = A(bf16 [M,1024]) @ Wt^T + bias ----------------
// Each wave computes a 32x64 output tile (2 A frags x 4 B frags = 8 WMMA accums),
// with register double-buffering of the next K-step's fragments so loads overlap WMMAs.
// mode 0: out bf16 [b,h,t,hd]   (Q, K)
// mode 1: out bf16 [b,h,hd,t]   (V transposed for P@V B-fragments)
// mode 2: out fp32 [M,1024]     (final projection -> d_out)
__global__ __launch_bounds__(128)
void gemm_bf16_wmma(const __bf16* __restrict__ A, const __bf16* __restrict__ Wt,
                    const float* __restrict__ bias, void* __restrict__ outp, int mode) {
  const int K = DM;
  const int MT = MR / 32;                       // 256 m-tiles of 32 rows
  int wave = (blockIdx.x * blockDim.x + threadIdx.x) >> 5;
  int lane = threadIdx.x & 31;
  int ln = lane & 15, hl = lane >> 4;
  int mt = wave % MT;
  int nh = wave / MT;                           // 64-column group == head index

  const __bf16* arow0 = A + (size_t)(mt * 32 + ln) * K + hl * 8;
  const __bf16* arow1 = arow0 + (size_t)16 * K;
  const __bf16* brow  = Wt + (size_t)(nh * 64 + ln) * K + hl * 8;

  v8f acc[2][4];
#pragma unroll
  for (int ms = 0; ms < 2; ++ms)
#pragma unroll
    for (int nt = 0; nt < 4; ++nt) acc[ms][nt] = (v8f){};

  v16bf aC[2], bC[4], aN[2], bN[4];
  aC[0] = cat8(*(const v8bf*)(arow0), *(const v8bf*)(arow0 + 16));
  aC[1] = cat8(*(const v8bf*)(arow1), *(const v8bf*)(arow1 + 16));
#pragma unroll
  for (int nt = 0; nt < 4; ++nt) bC[nt] = cat8(*(const v8bf*)(brow + (size_t)nt * 16 * K), *(const v8bf*)(brow + (size_t)nt * 16 * K + 16));

#pragma unroll 2
  for (int ks = 0; ks < K; ks += 32) {
    int kn = (ks + 32 < K) ? ks + 32 : 0;       // wrap: harmless dummy prefetch on last iter
    aN[0] = cat8(*(const v8bf*)(arow0 + kn), *(const v8bf*)(arow0 + kn + 16));
    aN[1] = cat8(*(const v8bf*)(arow1 + kn), *(const v8bf*)(arow1 + kn + 16));
#pragma unroll
    for (int nt = 0; nt < 4; ++nt) bN[nt] = cat8(*(const v8bf*)(brow + kn + (size_t)nt * 16 * K), *(const v8bf*)(brow + kn + 16 + (size_t)nt * 16 * K));
#pragma unroll
    for (int ms = 0; ms < 2; ++ms)
#pragma unroll
      for (int nt = 0; nt < 4; ++nt)
        acc[ms][nt] = wmma_bf16(aC[ms], bC[nt], acc[ms][nt]);
#pragma unroll
    for (int ms = 0; ms < 2; ++ms) aC[ms] = aN[ms];
#pragma unroll
    for (int nt = 0; nt < 4; ++nt) bC[nt] = bN[nt];
  }

#pragma unroll
  for (int nt = 0; nt < 4; ++nt) {
    float bs = bias[nh * 64 + nt * 16 + ln];
#pragma unroll
    for (int ms = 0; ms < 2; ++ms)
#pragma unroll
      for (int v = 0; v < 8; ++v) acc[ms][nt][v] += bs;
  }

  if (mode == 2) {                               // fp32 [M, 1024]
    float* out = (float*)outp;
#pragma unroll
    for (int ms = 0; ms < 2; ++ms)
#pragma unroll
      for (int nt = 0; nt < 4; ++nt)
#pragma unroll
        for (int v = 0; v < 8; ++v) {
          int row = mt * 32 + ms * 16 + hl * 8 + v;
          out[(size_t)row * DM + nh * 64 + nt * 16 + ln] = acc[ms][nt][v];
        }
  } else if (mode == 0) {                        // bf16 [b, h, t, hd]
    __bf16* out = (__bf16*)outp;
#pragma unroll
    for (int ms = 0; ms < 2; ++ms)
#pragma unroll
      for (int nt = 0; nt < 4; ++nt)
#pragma unroll
        for (int v = 0; v < 8; ++v) {
          int tg = mt * 32 + ms * 16 + hl * 8 + v;
          int b = tg / TS, t = tg & (TS - 1);
          size_t o = (((size_t)b * NH + nh) * TS + t) * HD + nt * 16 + ln;
          out[o] = (__bf16)acc[ms][nt][v];
        }
  } else {                                       // bf16 [b, h, hd, t] (V^T)
    __bf16* out = (__bf16*)outp;
#pragma unroll
    for (int ms = 0; ms < 2; ++ms) {
      int tg0 = mt * 32 + ms * 16 + hl * 8;
      int b = tg0 / TS, t0 = tg0 & (TS - 1);
#pragma unroll
      for (int nt = 0; nt < 4; ++nt) {
        v8bf pk;
#pragma unroll
        for (int v = 0; v < 8; ++v) pk[v] = (__bf16)acc[ms][nt][v];
        size_t o = (((size_t)b * NH + nh) * HD + nt * 16 + ln) * TS + t0;
        *(v8bf*)(out + o) = pk;                  // 8 contiguous t positions
      }
    }
  }
}

// ---------------- Flash attention (causal), wave = 16 query rows ----------------
__global__ __launch_bounds__(128)
void attn_wmma(const __bf16* __restrict__ Q, const __bf16* __restrict__ Kb,
               const __bf16* __restrict__ Vt, __bf16* __restrict__ ctx) {
  __shared__ __align__(16) __bf16 plds[4][16 * 32];  // per-wave P transpose staging
  int widx = threadIdx.x >> 5;
  int lane = threadIdx.x & 31;
  int ln = lane & 15, hl = lane >> 4;
  int wave = blockIdx.x * 4 + widx;
  const int QT = TS / 16;                        // 128 q-tiles per head
  int qi = wave % QT;
  int bh = wave / QT;                            // 0..63

  const __bf16* Qh = Q  + (size_t)bh * TS * HD;
  const __bf16* Kh = Kb + (size_t)bh * TS * HD;
  const __bf16* Vh = Vt + (size_t)bh * HD * TS;

  // Q fragments (16 rows x 64 d, two K-steps of 32)
  const __bf16* qrow = Qh + (size_t)(qi * 16 + ln) * HD;
  v16bf qf[2];
#pragma unroll
  for (int s = 0; s < 2; ++s)
    qf[s] = cat8(*(const v8bf*)(qrow + s * 32 + hl * 8),
                 *(const v8bf*)(qrow + s * 32 + 16 + hl * 8));

  float rmax[8], rsum[8];
  v8f acc[4];
#pragma unroll
  for (int v = 0; v < 8; ++v) { rmax[v] = -3.0e38f; rsum[v] = 0.0f; }
#pragma unroll
  for (int nt = 0; nt < 4; ++nt) acc[nt] = (v8f){};

  const float scale = 0.125f;                    // 1/sqrt(64)
  int qbase = qi * 16;
  int nb = (qbase + 16 + 31) / 32;               // causal key-block bound
  __bf16* pl = plds[widx];

  auto loadK = [&](int kb, v16bf kf[2][2]) {
#pragma unroll
    for (int st2 = 0; st2 < 2; ++st2) {
      const __bf16* kr = Kh + (size_t)(kb + st2 * 16 + ln) * HD + hl * 8;
      kf[st2][0] = cat8(*(const v8bf*)(kr), *(const v8bf*)(kr + 16));
      kf[st2][1] = cat8(*(const v8bf*)(kr + 32), *(const v8bf*)(kr + 48));
    }
  };

  v16bf kfC[2][2], kfN[2][2];
  loadK(0, kfC);

  for (int jb = 0; jb < nb; ++jb) {
    int kb = jb * 32;
    // ---- scores: two 16x16 tiles over 32 keys ----
    v8f s[2];
#pragma unroll
    for (int st2 = 0; st2 < 2; ++st2) {
      v8f sc = (v8f){};
      sc = wmma_bf16(qf[0], kfC[st2][0], sc);
      sc = wmma_bf16(qf[1], kfC[st2][1], sc);
      s[st2] = sc;
    }
    // prefetch next key block's K fragments while softmax runs
    int kpre = (jb + 1 < nb) ? kb + 32 : 0;
    loadK(kpre, kfN);
    // issue V fragment loads early (independent of P)
    v16bf vf[4];
#pragma unroll
    for (int nt = 0; nt < 4; ++nt)
      vf[nt] = cat8(*(const v8bf*)(Vh + (size_t)(nt * 16 + ln) * TS + kb + hl * 8), *(const v8bf*)(Vh + (size_t)(nt * 16 + ln) * TS + kb + 16 + hl * 8));

    // ---- online softmax (rows live across 16 lanes of each half) ----
    float x0[8], x1[8], cand[8];
#pragma unroll
    for (int v = 0; v < 8; ++v) {
      int qg = qbase + hl * 8 + v;
      float a = s[0][v] * scale; if (kb + ln > qg)      a = -3.0e38f;
      float b = s[1][v] * scale; if (kb + 16 + ln > qg) b = -3.0e38f;
      x0[v] = a; x1[v] = b; cand[v] = fmaxf(a, b);
    }
#pragma unroll
    for (int m = 1; m < 16; m <<= 1)
#pragma unroll
      for (int v = 0; v < 8; ++v) cand[v] = fmaxf(cand[v], __shfl_xor(cand[v], m, 32));

    float alpha[8], rows[8], p0[8], p1[8];
#pragma unroll
    for (int v = 0; v < 8; ++v) {
      float mnew = fmaxf(rmax[v], cand[v]);
      alpha[v] = __expf(rmax[v] - mnew);
      p0[v] = __expf(x0[v] - mnew);
      p1[v] = __expf(x1[v] - mnew);
      rmax[v] = mnew;
      rows[v] = p0[v] + p1[v];
    }
#pragma unroll
    for (int m = 1; m < 16; m <<= 1)
#pragma unroll
      for (int v = 0; v < 8; ++v) rows[v] += __shfl_xor(rows[v], m, 32);
#pragma unroll
    for (int v = 0; v < 8; ++v) rsum[v] = rsum[v] * alpha[v] + rows[v];
#pragma unroll
    for (int nt = 0; nt < 4; ++nt)
#pragma unroll
      for (int v = 0; v < 8; ++v) acc[nt][v] *= alpha[v];

    // ---- P: C-layout -> A-layout via LDS bounce ----
#pragma unroll
    for (int v = 0; v < 8; ++v) {
      pl[(hl * 8 + v) * 32 + ln]      = (__bf16)p0[v];
      pl[(hl * 8 + v) * 32 + 16 + ln] = (__bf16)p1[v];
    }
    asm volatile("s_wait_dscnt 0" ::: "memory");
    v16bf pf = cat8(*(const v8bf*)(pl + ln * 32 + hl * 8),
                    *(const v8bf*)(pl + ln * 32 + 16 + hl * 8));

    // ---- ctx += P @ V  (V stored transposed [d][t]) ----
#pragma unroll
    for (int nt = 0; nt < 4; ++nt)
      acc[nt] = wmma_bf16(pf, vf[nt], acc[nt]);

#pragma unroll
    for (int st2 = 0; st2 < 2; ++st2) {
      kfC[st2][0] = kfN[st2][0];
      kfC[st2][1] = kfN[st2][1];
    }
  }

  // epilogue: ctx[b, t, h*64 + d] bf16, row-major [M, 1024] for output GEMM
  int b = bh / NH, h = bh % NH;
#pragma unroll
  for (int nt = 0; nt < 4; ++nt)
#pragma unroll
    for (int v = 0; v < 8; ++v) {
      int t = qbase + hl * 8 + v;
      size_t o = ((size_t)(b * TS + t)) * DM + h * 64 + nt * 16 + ln;
      ctx[o] = (__bf16)(acc[nt][v] / rsum[v]);
    }
}

extern "C" void kernel_launch(void* const* d_in, const int* in_sizes, int n_in,
                              void* d_out, int out_size, void* d_ws, size_t ws_size,
                              hipStream_t stream) {
  const float* x  = (const float*)d_in[0];
  const float* Wq = (const float*)d_in[1];
  const float* bq = (const float*)d_in[2];
  const float* Wk = (const float*)d_in[3];
  const float* bk = (const float*)d_in[4];
  const float* Wv = (const float*)d_in[5];
  const float* bv = (const float*)d_in[6];
  const float* Wo = (const float*)d_in[7];
  const float* bo = (const float*)d_in[8];

  char* ws = (char*)d_ws;
  __bf16* xb  = (__bf16*)(ws);                         // 16 MB
  __bf16* Wqt = (__bf16*)(ws + (16ull << 20));         //  2 MB
  __bf16* Wkt = (__bf16*)(ws + (18ull << 20));
  __bf16* Wvt = (__bf16*)(ws + (20ull << 20));
  __bf16* Wot = (__bf16*)(ws + (22ull << 20));
  __bf16* Qb  = (__bf16*)(ws + (24ull << 20));         // 16 MB [b,h,t,hd]
  __bf16* Kbf = (__bf16*)(ws + (40ull << 20));         // 16 MB [b,h,t,hd]
  __bf16* Vtb = (__bf16*)(ws + (56ull << 20));         // 16 MB [b,h,hd,t]
  __bf16* Cb  = (__bf16*)(ws + (72ull << 20));         // 16 MB [M, 1024]

  cvt_kernel<<<4096, 256, 0, stream>>>(x, xb, MR * DM);
  transw_kernel<<<4096, 256, 0, stream>>>(Wq, Wqt);
  transw_kernel<<<4096, 256, 0, stream>>>(Wk, Wkt);
  transw_kernel<<<4096, 256, 0, stream>>>(Wv, Wvt);
  transw_kernel<<<4096, 256, 0, stream>>>(Wo, Wot);

  // 256 m-tiles (32 rows) * 16 n-groups = 4096 waves, 4 waves/block
  gemm_bf16_wmma<<<1024, 128, 0, stream>>>(xb, Wqt, bq, Qb, 0);
  gemm_bf16_wmma<<<1024, 128, 0, stream>>>(xb, Wkt, bk, Kbf, 0);
  gemm_bf16_wmma<<<1024, 128, 0, stream>>>(xb, Wvt, bv, Vtb, 1);

  // 64 (b,h) * 128 q-tiles = 8192 waves
  attn_wmma<<<2048, 128, 0, stream>>>(Qb, Kbf, Vtb, Cb);

  gemm_bf16_wmma<<<1024, 128, 0, stream>>>(Cb, Wot, bo, d_out, 2);
}
